// TransformerDecoder_15857019257514
// MI455X (gfx1250) — matrix operand order repaired and checked by reading, not compiled, not run on hardware
//
#include <hip/hip_runtime.h>
#include <hip/hip_bf16.h>
#include <stdint.h>

// Problem constants (match reference)
#define B_  4
#define TT_ 512
#define TM_ 512
#define D_  1024
#define H_  16
#define HD_ 64
#define L_  6
#define F_  4096
#define V_  32000

typedef __bf16 v16bf __attribute__((ext_vector_type(16)));
typedef float  v8f   __attribute__((ext_vector_type(8)));

union FragAB { v16bf v; unsigned u[8]; };
union FragC  { v8f v; float f[8]; };

__device__ __forceinline__ unsigned short f2bf(float f) {
  union { float f; unsigned u; } c; c.f = f;
  return (unsigned short)((c.u + 0x7FFFu + ((c.u >> 16) & 1u)) >> 16);  // RNE
}
__device__ __forceinline__ unsigned pack2(float a, float b) {
  return (unsigned)f2bf(a) | ((unsigned)f2bf(b) << 16);
}

// ---------------------------------------------------------------------------
// Generic GEMM: C[M,N] = A[M,K] @ W[K,N] + bias (+ReLU) (+residual R)
// Block tile 128x128x32, 256 threads = 8 wave32 (2x4), wave tile 64x32.
// Double-buffered LDS + register prefetch: one barrier per k-step, next
// tile's global loads (global_load_b128) overlap the current 8 WMMAs.
// flags: bit0 = ReLU, bit1 = add residual R[M,N]
// ---------------------------------------------------------------------------
__global__ __launch_bounds__(256)
void gemm_bias_kernel(const float* __restrict__ A, const float* __restrict__ W,
                      const float* __restrict__ bias, const float* __restrict__ R,
                      float* __restrict__ C, int M, int N, int K, int flags) {
  __shared__ unsigned Asm[2][128 * 17];  // [row][k/2] bf16x2, stride 17 dwords
  __shared__ unsigned Bsm[2][128 * 17];  // [col][k/2] bf16x2 (n-major, k-contiguous)

  const int m0 = blockIdx.y * 128;
  const int n0 = blockIdx.x * 128;
  const int t = threadIdx.x;
  const int lane = t & 31, wave = t >> 5;
  const int wm = (wave >> 2) * 64;   // wave M offset
  const int wn = (wave & 3) * 32;    // wave N offset
  const int r = lane & 15, half = lane >> 4;

  // staging coordinates (per thread: A = 16 consecutive floats of one row,
  // B = 16 consecutive floats of one W row, scattered transposed into LDS)
  const int arow = t >> 1, acb = (t & 1) * 16;
  const int bkk = t >> 3, bnb = (t & 7) * 16;
  const float* aSrc = A + (size_t)(m0 + arow) * K + acb;   // + s*32 per step
  const float* bSrc = W + (size_t)bkk * N + n0 + bnb;      // + s*32*N per step

  FragC acc[4][2];
#pragma unroll
  for (int mi = 0; mi < 4; ++mi)
#pragma unroll
    for (int ni = 0; ni < 2; ++ni)
#pragma unroll
      for (int e = 0; e < 8; ++e) acc[mi][ni].f[e] = 0.f;

  float4 aPre[4], bPre[4];
  auto prefetch = [&](int s) {
    const float4* pa = (const float4*)(aSrc + s * 32);
    aPre[0] = pa[0]; aPre[1] = pa[1]; aPre[2] = pa[2]; aPre[3] = pa[3];
    const float4* pb = (const float4*)(bSrc + (size_t)s * 32 * N);
    bPre[0] = pb[0]; bPre[1] = pb[1]; bPre[2] = pb[2]; bPre[3] = pb[3];
  };
  auto stage = [&](int buf) {
    unsigned* ad = Asm[buf] + arow * 17 + (acb >> 1);
#pragma unroll
    for (int j = 0; j < 4; ++j) {
      ad[2 * j]     = pack2(aPre[j].x, aPre[j].y);
      ad[2 * j + 1] = pack2(aPre[j].z, aPre[j].w);
    }
    unsigned short* bs = (unsigned short*)Bsm[buf];
#pragma unroll
    for (int j = 0; j < 4; ++j) {
      bs[(bnb + 4 * j + 0) * 34 + bkk] = f2bf(bPre[j].x);
      bs[(bnb + 4 * j + 1) * 34 + bkk] = f2bf(bPre[j].y);
      bs[(bnb + 4 * j + 2) * 34 + bkk] = f2bf(bPre[j].z);
      bs[(bnb + 4 * j + 3) * 34 + bkk] = f2bf(bPre[j].w);
    }
  };

  prefetch(0);
  stage(0);
  __syncthreads();

  const int nk = K >> 5;
  for (int s = 0; s < nk; ++s) {
    const int cur = s & 1;
    if (s + 1 < nk) prefetch(s + 1);   // global loads overlap WMMAs below

    FragAB a[4], bfr[2];
#pragma unroll
    for (int mi = 0; mi < 4; ++mi) {
      // A 16x32 layout: lanes0-15 K in [0,8)+[16,24), lanes16-31 +8
      const unsigned* p = Asm[cur] + (wm + mi * 16 + r) * 17 + half * 4;
#pragma unroll
      for (int j = 0; j < 8; ++j) a[mi].u[j] = (j < 4) ? p[j] : p[8 + (j - 4)];
    }
#pragma unroll
    for (int ni = 0; ni < 2; ++ni) {
      // B 32x16 layout: lanes0-15 K in [0,8)+[16,24), lanes16-31 +8
      const unsigned* p = Bsm[cur] + (wn + ni * 16 + r) * 17 + half * 4;
#pragma unroll
      for (int j = 0; j < 8; ++j) bfr[ni].u[j] = (j < 4) ? p[j] : p[8 + (j - 4)];
    }
#pragma unroll
    for (int mi = 0; mi < 4; ++mi)
#pragma unroll
      for (int ni = 0; ni < 2; ++ni)
        acc[mi][ni].v = __builtin_amdgcn_wmma_f32_16x16x32_bf16(
            false, a[mi].v, false, bfr[ni].v, (short)0, acc[mi][ni].v, false, false);

    if (s + 1 < nk) stage(cur ^ 1);    // write other buffer: no race with readers
    __syncthreads();                   // single barrier per k-step
  }

  // Epilogue: C layout VGPR e -> M = e (lanes0-15) / e+8 (lanes16-31), N = lane&15
#pragma unroll
  for (int mi = 0; mi < 4; ++mi)
#pragma unroll
    for (int ni = 0; ni < 2; ++ni)
#pragma unroll
      for (int e = 0; e < 8; ++e) {
        int row = m0 + wm + mi * 16 + e + 8 * half;
        int col = n0 + wn + ni * 16 + r;
        float vv = acc[mi][ni].f[e] + bias[col];
        if (flags & 1) vv = fmaxf(vv, 0.f);
        size_t idx = (size_t)row * N + col;
        if (flags & 2) vv += R[idx];
        C[idx] = vv;
      }
}

// ---------------------------------------------------------------------------
// LayerNorm: one block per row
// ---------------------------------------------------------------------------
__global__ __launch_bounds__(256)
void layernorm_kernel(const float* __restrict__ x, const float* __restrict__ g,
                      const float* __restrict__ b, float* __restrict__ y, int D) {
  __shared__ float red[256];
  const int row = blockIdx.x;
  const float* xr = x + (size_t)row * D;
  float* yr = y + (size_t)row * D;
  const int t = threadIdx.x;

  float s = 0.f;
  for (int i = t; i < D; i += 256) s += xr[i];
  red[t] = s; __syncthreads();
  for (int off = 128; off > 0; off >>= 1) {
    if (t < off) red[t] += red[t + off];
    __syncthreads();
  }
  float mu = red[0] / (float)D; __syncthreads();

  float s2 = 0.f;
  for (int i = t; i < D; i += 256) { float d = xr[i] - mu; s2 += d * d; }
  red[t] = s2; __syncthreads();
  for (int off = 128; off > 0; off >>= 1) {
    if (t < off) red[t] += red[t + off];
    __syncthreads();
  }
  float inv = rsqrtf(red[0] / (float)D + 1e-5f);
  for (int i = t; i < D; i += 256) yr[i] = (xr[i] - mu) * inv * g[i] + b[i];
}

// ---------------------------------------------------------------------------
// Embedding gather
// ---------------------------------------------------------------------------
__global__ __launch_bounds__(256)
void embed_kernel(const int* __restrict__ ids, const float* __restrict__ emb,
                  float* __restrict__ x) {
  const int row = blockIdx.x;
  const int id = ids[row];
  const float* src = emb + (size_t)id * D_;
  float* dst = x + (size_t)row * D_;
  for (int i = threadIdx.x; i < D_; i += 256) dst[i] = src[i];
}

// ---------------------------------------------------------------------------
// Fused attention: one block per (b, h, 16-query tile). Tk fixed at 512.
// Phase1: S = scale*Q K^T (WMMA, wave owns 64 keys), masked into LDS fp32.
// Phase2: row softmax fp32 in LDS, pack P to bf16.
// Phase3: O = P@V (WMMA, split-K over 8 waves) + LDS reduction.
// mode 0: causal + key<lens[b];  mode 1: memmask[b*Tk+key]!=0
// ---------------------------------------------------------------------------
__global__ __launch_bounds__(256)
void attention_kernel(const float* __restrict__ Q, const float* __restrict__ Km,
                      const float* __restrict__ Vm, float* __restrict__ O,
                      int Tq, int Tk, int Dm, int nH, int mode,
                      const int* __restrict__ lens, const int* __restrict__ memmask,
                      float scale) {
  __shared__ float Ssm[16 * 512];      // scores fp32; reused as O partials (8x16x64)
  __shared__ unsigned Psm[16 * 257];   // P bf16x2, row stride 257 dwords (514 halves)
  __shared__ float rowred[16][17];

  const int nQT = Tq >> 4;
  const int bx = blockIdx.x;
  const int qt = bx % nQT;
  const int h = (bx / nQT) % nH;
  const int b = bx / (nQT * nH);

  const int t = threadIdx.x;
  const int lane = t & 31, wave = t >> 5;
  const int r = lane & 15, half = lane >> 4;

  const size_t hoff = (size_t)h * HD_;
  const float* Qb = Q + ((size_t)(b * Tq + qt * 16)) * Dm + hoff;
  const float* Kb = Km + ((size_t)b * Tk) * Dm + hoff;
  const float* Vb = Vm + ((size_t)b * Tk) * Dm + hoff;

  // ---- Phase 1: scores; this wave owns keys [wave*64, wave*64+64) ----
  FragC acc[4];
#pragma unroll
  for (int ni = 0; ni < 4; ++ni)
#pragma unroll
    for (int e = 0; e < 8; ++e) acc[ni].f[e] = 0.f;

  const int kw = wave * 64;
  for (int kk = 0; kk < HD_; kk += 32) {
    FragAB a;
    const float* qrow = Qb + (size_t)r * Dm + kk + half * 8;
#pragma unroll
    for (int j = 0; j < 8; ++j) {
      int off = (j < 4) ? 2 * j : 16 + 2 * (j - 4);
      float2 f2 = *(const float2*)(qrow + off);
      a.u[j] = pack2(f2.x, f2.y);
    }
#pragma unroll
    for (int ni = 0; ni < 4; ++ni) {
      FragAB bfr;
      const float* krow = Kb + (size_t)(kw + ni * 16 + r) * Dm + kk + half * 8;
#pragma unroll
      for (int j = 0; j < 8; ++j) {
        float2 f2 = *(const float2*)(krow + ((j < 4) ? 2 * j : 16 + 2 * (j - 4)));
        bfr.u[j] = pack2(f2.x, f2.y);
      }
      acc[ni].v = __builtin_amdgcn_wmma_f32_16x16x32_bf16(
          false, a.v, false, bfr.v, (short)0, acc[ni].v, false, false);
    }
  }
  const int lim = (mode == 0) ? lens[b] : 0;
#pragma unroll
  for (int ni = 0; ni < 4; ++ni)
#pragma unroll
    for (int e = 0; e < 8; ++e) {
      int ql = e + 8 * half;
      int key = kw + ni * 16 + r;
      bool ok;
      if (mode == 0) ok = (key < lim) && (key <= qt * 16 + ql);
      else ok = (memmask[(size_t)b * Tk + key] != 0);
      Ssm[ql * 512 + key] = ok ? acc[ni].f[e] * scale : -1e9f;
    }
  __syncthreads();

  // ---- Phase 2: softmax over each of the 16 rows (16 threads per row) ----
  {
    int row = t >> 4, c0 = t & 15;
    float mx = -3.0e38f;
    for (int c = c0; c < 512; c += 16) mx = fmaxf(mx, Ssm[row * 512 + c]);
    rowred[row][c0] = mx; __syncthreads();
    if (c0 == 0) {
      float m = rowred[row][0];
      for (int i = 1; i < 16; ++i) m = fmaxf(m, rowred[row][i]);
      rowred[row][16] = m;
    }
    __syncthreads();
    mx = rowred[row][16];
    float sum = 0.f;
    for (int c = c0; c < 512; c += 16) {
      float e = __expf(Ssm[row * 512 + c] - mx);
      Ssm[row * 512 + c] = e;
      sum += e;
    }
    __syncthreads();
    rowred[row][c0] = sum; __syncthreads();
    if (c0 == 0) {
      float s2 = 0.f;
      for (int i = 0; i < 16; ++i) s2 += rowred[row][i];
      rowred[row][16] = s2;
    }
    __syncthreads();
    float inv = 1.0f / rowred[row][16];
    for (int p = c0; p < 256; p += 16)
      Psm[row * 257 + p] =
          pack2(Ssm[row * 512 + 2 * p] * inv, Ssm[row * 512 + 2 * p + 1] * inv);
  }
  __syncthreads();

  // ---- Phase 3: O = P @ V, split-K over waves (each wave: 64 keys) ----
  FragC oacc[4];
#pragma unroll
  for (int ni = 0; ni < 4; ++ni)
#pragma unroll
    for (int e = 0; e < 8; ++e) oacc[ni].f[e] = 0.f;

#pragma unroll
  for (int s = 0; s < 2; ++s) {
    int kk = wave * 64 + s * 32;
    FragAB a;
    const unsigned* prow = Psm + r * 257 + (kk >> 1) + half * 4;
#pragma unroll
    for (int j = 0; j < 8; ++j) a.u[j] = (j < 4) ? prow[j] : prow[8 + (j - 4)];
#pragma unroll
    for (int ni = 0; ni < 4; ++ni) {
      FragAB bfr;
      int d = ni * 16 + r;
#pragma unroll
      for (int j = 0; j < 8; ++j) {
        int key = kk + half * 8 + ((j < 4) ? 2 * j : 16 + 2 * (j - 4));
        float f0 = Vb[(size_t)key * Dm + d];
        float f1 = Vb[(size_t)(key + 1) * Dm + d];
        bfr.u[j] = pack2(f0, f1);
      }
      oacc[ni].v = __builtin_amdgcn_wmma_f32_16x16x32_bf16(
          false, a.v, false, bfr.v, (short)0, oacc[ni].v, false, false);
    }
  }
  // stash partials in Ssm (free after softmax), then sum the 8 partials across waves
#pragma unroll
  for (int ni = 0; ni < 4; ++ni)
#pragma unroll
    for (int e = 0; e < 8; ++e) {
      int ql = e + 8 * half;
      int d = ni * 16 + r;
      Ssm[wave * 1024 + ql * 64 + d] = oacc[ni].f[e];
    }
  __syncthreads();
  float* Ob = O + ((size_t)(b * Tq + qt * 16)) * Dm + hoff;
  for (int o = t; o < 1024; o += 256) {
    int ql = o >> 6, d = o & 63;
    float s2 = 0.f;
#pragma unroll
    for (int w = 0; w < 8; ++w) s2 += Ssm[w * 1024 + o];
    Ob[(size_t)ql * Dm + d] = s2;
  }
}

// ---------------------------------------------------------------------------
// Host orchestration
// ---------------------------------------------------------------------------
extern "C" void kernel_launch(void* const* d_in, const int* in_sizes, int n_in,
                              void* d_out, int out_size, void* d_ws, size_t ws_size,
                              hipStream_t stream) {
  (void)in_sizes; (void)n_in; (void)out_size; (void)ws_size;
  const float* memory  = (const float*)d_in[0];
  const int* memmask   = (const int*)d_in[1];
  const int* ys_pad    = (const int*)d_in[2];
  const int* ys_lens   = (const int*)d_in[3];
  const float* embed   = (const float*)d_in[4];
  const float* Wq_s = (const float*)d_in[5];  const float* bq_s = (const float*)d_in[6];
  const float* Wk_s = (const float*)d_in[7];  const float* bk_s = (const float*)d_in[8];
  const float* Wv_s = (const float*)d_in[9];  const float* bv_s = (const float*)d_in[10];
  const float* Wo_s = (const float*)d_in[11]; const float* bo_s = (const float*)d_in[12];
  const float* Wq_c = (const float*)d_in[13]; const float* bq_c = (const float*)d_in[14];
  const float* Wk_c = (const float*)d_in[15]; const float* bk_c = (const float*)d_in[16];
  const float* Wv_c = (const float*)d_in[17]; const float* bv_c = (const float*)d_in[18];
  const float* Wo_c = (const float*)d_in[19]; const float* bo_c = (const float*)d_in[20];
  const float* W1 = (const float*)d_in[21];   const float* b1 = (const float*)d_in[22];
  const float* W2 = (const float*)d_in[23];   const float* b2 = (const float*)d_in[24];
  const float* ln1_g = (const float*)d_in[25]; const float* ln1_b = (const float*)d_in[26];
  const float* ln2_g = (const float*)d_in[27]; const float* ln2_b = (const float*)d_in[28];
  const float* ln3_g = (const float*)d_in[29]; const float* ln3_b = (const float*)d_in[30];
  const float* Wout = (const float*)d_in[31];  const float* bout = (const float*)d_in[32];

  const int M = B_ * TT_;              // 2048
  const size_t MD = (size_t)M * D_;
  float* x  = (float*)d_ws;
  float* xn = x + MD;
  float* mn = xn + MD;
  float* q  = mn + MD;
  float* k  = q + MD;
  float* v  = k + MD;
  float* o  = v + MD;
  float* hb = o + MD;                  // 2048 x 4096

  auto gemm = [&](const float* A, const float* W, const float* bias, const float* R,
                  float* C, int Mr, int Nc, int Kc, int flags) {
    dim3 g(Nc / 128, Mr / 128);
    gemm_bias_kernel<<<g, 256, 0, stream>>>(A, W, bias, R, C, Mr, Nc, Kc, flags);
  };

  embed_kernel<<<M, 256, 0, stream>>>(ys_pad, embed, x);

  for (int l = 0; l < L_; ++l) {
    size_t w_off = (size_t)l * D_ * D_;
    size_t v_off = (size_t)l * D_;
    size_t ff_off = (size_t)l * D_ * F_;
    size_t fb_off = (size_t)l * F_;

    // self-attention block
    layernorm_kernel<<<M, 256, 0, stream>>>(x, ln1_g + v_off, ln1_b + v_off, xn, D_);
    gemm(xn, Wq_s + w_off, bq_s + v_off, nullptr, q, M, D_, D_, 0);
    gemm(xn, Wk_s + w_off, bk_s + v_off, nullptr, k, M, D_, D_, 0);
    gemm(xn, Wv_s + w_off, bv_s + v_off, nullptr, v, M, D_, D_, 0);
    attention_kernel<<<B_ * H_ * (TT_ / 16), 256, 0, stream>>>(
        q, k, v, o, TT_, TT_, D_, H_, 0, ys_lens, nullptr, 0.125f);
    gemm(o, Wo_s + w_off, bo_s + v_off, x, x, M, D_, D_, 2);   // residual

    // cross-attention block (memory normed with ln2 as in reference)
    layernorm_kernel<<<M, 256, 0, stream>>>(x, ln2_g + v_off, ln2_b + v_off, xn, D_);
    layernorm_kernel<<<B_ * TM_, 256, 0, stream>>>(memory, ln2_g + v_off, ln2_b + v_off, mn, D_);
    gemm(xn, Wq_c + w_off, bq_c + v_off, nullptr, q, M, D_, D_, 0);
    gemm(mn, Wk_c + w_off, bk_c + v_off, nullptr, k, B_ * TM_, D_, D_, 0);
    gemm(mn, Wv_c + w_off, bv_c + v_off, nullptr, v, B_ * TM_, D_, D_, 0);
    attention_kernel<<<B_ * H_ * (TT_ / 16), 256, 0, stream>>>(
        q, k, v, o, TT_, TM_, D_, H_, 1, nullptr, memmask, 0.125f);
    gemm(o, Wo_c + w_off, bo_c + v_off, x, x, M, D_, D_, 2);   // residual

    // FFN
    layernorm_kernel<<<M, 256, 0, stream>>>(x, ln3_g + v_off, ln3_b + v_off, xn, D_);
    gemm(xn, W1 + ff_off, b1 + fb_off, nullptr, hb, M, F_, D_, 1);  // ReLU
    gemm(hb, W2 + ff_off, b2 + v_off, x, x, M, D_, F_, 2);          // residual
  }

  // final vocab projection: [2048,1024] @ [1024,32000] -> d_out
  gemm(x, Wout, bout, nullptr, (float*)d_out, M, V_, D_, 0);
}
